// WTFTP_AttnRemoved_76596446757349
// MI455X (gfx1250) — matrix operand order repaired and checked by reading, not compiled, not run on hardware
//
#include <hip/hip_runtime.h>
#include <hip/hip_bf16.h>

typedef __attribute__((ext_vector_type(16))) __bf16 v16bf;
typedef __attribute__((ext_vector_type(8)))  __bf16 v8bf;
typedef __attribute__((ext_vector_type(8)))  float  v8f;
typedef __attribute__((ext_vector_type(4)))  unsigned int ui32x4;
typedef __attribute__((ext_vector_type(8)))  int i32x8;
typedef __attribute__((ext_vector_type(4)))  int i32x4;

#define HID   128
#define TLEN  64
#define BATCH 2048
#define BT    32              // batch rows per workgroup
#define GATES 512             // 4*HID
#define KTOT  256             // 2*HID (x ‖ h)
#define WELEM (GATES*KTOT)    // 131072 bf16 per fused layer weight
#define WBYTES ((size_t)WELEM*2)
#define XBYTES ((size_t)BT*HID*2)

#if defined(__has_builtin)
# if __has_builtin(__builtin_amdgcn_tensor_load_to_lds)
#  define HAVE_TDM 1
# endif
#endif
#ifndef HAVE_TDM
# define HAVE_TDM 0
#endif

#if HAVE_TDM
// 2-D Tensor-Data-Mover load: tile == tensor window, element size 2 bytes.
// D# per CDNA5 ISA 8.3/8.4: group0 {count, lds_addr, global_addr, type=2},
// group1 {data_size=1(2B), tensor_dim0/1, tile_dim0/1, tensor_dim0_stride}.
// This toolchain exposes the 6-arg builtin: (g0, g1, g2, g3, g4, cpol).
__device__ __forceinline__ void tdm_load_2d(unsigned lds_off, const void* gptr,
                                            unsigned dim0, unsigned dim1,
                                            unsigned stride0 /*elements*/){
  unsigned long long ga = (unsigned long long)(size_t)gptr;
  ui32x4 g0;
  g0[0] = 1u;                                            // count=1, user mode
  g0[1] = lds_off;                                       // LDS byte address
  g0[2] = (unsigned)(ga & 0xFFFFFFFFull);                // global addr [31:0]
  g0[3] = (unsigned)((ga >> 32) & 0x1FFFFFFull) | (2u << 30); // [56:32] | type=2
  i32x8 g1;
  g1[0] = (int)(1u << 16);                               // data_size=2B, mask=0
  g1[1] = (int)(dim0 << 16);                             // tensor_dim0[15:0]
  g1[2] = (int)((dim0 >> 16) | (dim1 << 16));            // dim0 hi | tensor_dim1 lo
  g1[3] = (int)((dim1 >> 16) | (dim0 << 16));            // dim1 hi | tile_dim0
  g1[4] = (int)(dim1 & 0xFFFFu);                         // tile_dim1 (tile_dim2=0)
  g1[5] = (int)stride0;                                  // tensor_dim0_stride lo32
  g1[6] = 0;
  g1[7] = 0;
  i32x4 z4 = {0,0,0,0};                                  // groups 2/3 unused (2-D)
  i32x8 z8 = {0,0,0,0,0,0,0,0};
  __builtin_amdgcn_tensor_load_to_lds(g0, g1, z4, z4, z8, 0);
}
#endif

// ---------------------------------------------------------------- fragments
__device__ __forceinline__ v16bf cat16(v8bf lo, v8bf hi){
  return __builtin_shufflevector(lo, hi, 0,1,2,3,4,5,6,7,8,9,10,11,12,13,14,15);
}
// A-matrix 16x32 bf16 (ISA 7.12.2): lanes 0-15 rows M, K 0-7 & 16-23; lanes 16-31 K 8-15 & 24-31
__device__ __forceinline__ v16bf load_afrag(const __bf16* base, int mtile, int kk, int lane){
  const int mm = lane & 15, hf = lane >> 4;
  const __bf16* p = base + (mtile*16 + mm)*HID + kk*32 + hf*8;
  v8bf lo = *(const v8bf*)(p);
  v8bf hi = *(const v8bf*)(p + 16);
  return cat16(lo, hi);
}
// B-matrix 32x16 bf16: lane = column N, lanes 0-15 hold K 0-7 & 16-23, lanes 16-31 hold K 8-15 & 24-31
__device__ __forceinline__ v16bf load_bfrag(const __bf16* W, int ntile, int kk, int lane){
  const int nn = lane & 15, hf = lane >> 4;
  const __bf16* p = W + (ntile*16 + nn)*KTOT + kk*32 + hf*8;
  v8bf lo = *(const v8bf*)(p);
  v8bf hi = *(const v8bf*)(p + 16);
  return cat16(lo, hi);
}
__device__ __forceinline__ v8f wmma_bf16(v16bf a, v16bf b, v8f c){
  return __builtin_amdgcn_wmma_f32_16x16x32_bf16(false, a, false, b, (short)0, c, false, false);
}
__device__ __forceinline__ float sigm(float x){ return 1.0f/(1.0f+__expf(-x)); }
__device__ __forceinline__ float tanh_f(float x){ return 1.0f - 2.0f/(__expf(2.0f*x)+1.0f); }

// One fused LSTM-cell GEMM: gates(32x512) = [Xb|Hb](32x256) @ Wlds^T(256x512)
__device__ __forceinline__ void lstm_cell_gemm(const __bf16* Wlds, const __bf16* Xb,
                                               const __bf16* Hb, int j, int lane,
                                               v8f acc[2][4]){
#pragma unroll
  for (int m=0;m<2;m++)
#pragma unroll
    for (int g=0;g<4;g++)
#pragma unroll
      for (int r=0;r<8;r++) acc[m][g][r]=0.0f;
#pragma unroll
  for (int kk=0; kk<8; kk++){
    const __bf16* asrc = (kk<4) ? Xb : Hb;
    const int kx = kk & 3;
    v16bf a0 = load_afrag(asrc, 0, kx, lane);
    v16bf a1 = load_afrag(asrc, 1, kx, lane);
#pragma unroll
    for (int g=0; g<4; g++){
      v16bf b = load_bfrag(Wlds, g*8 + j, kk, lane);
      acc[0][g] = wmma_bf16(a0, b, acc[0][g]);
      acc[1][g] = wmma_bf16(a1, b, acc[1][g]);
    }
  }
}

// ---------------------------------------------------------------- prologue kernels
__global__ __launch_bounds__(256) void prep_weights_kernel(
    const float* __restrict__ eWih, const float* __restrict__ eWhh,
    const float* __restrict__ dWih, const float* __restrict__ dWhh,
    __bf16* __restrict__ WcE, __bf16* __restrict__ WcD){
  const int encN = 4*WELEM, decN = 8*WELEM;
  int idx = blockIdx.x*256 + threadIdx.x;
  if (idx < encN){
    int l = idx/WELEM, rem = idx%WELEM, n = rem/KTOT, k = rem%KTOT;
    float v = (k<HID) ? eWih[((size_t)l*GATES+n)*HID + k]
                      : eWhh[((size_t)l*GATES+n)*HID + (k-HID)];
    WcE[idx] = (__bf16)v;
  } else if (idx < encN + decN){
    int q = idx - encN;
    int dl = q/WELEM, rem = q%WELEM, n = rem/KTOT, k = rem%KTOT;
    float v = (k<HID) ? dWih[((size_t)dl*GATES+n)*HID + k]
                      : dWhh[((size_t)dl*GATES+n)*HID + (k-HID)];
    WcD[q] = (__bf16)v;
  }
}

__global__ __launch_bounds__(128) void embed_kernel(
    const float* __restrict__ inp, const float* __restrict__ W1,
    const float* __restrict__ W2, __bf16* __restrict__ x0){
  __shared__ float W2s[HID*64];
  __shared__ float e[64];
  __shared__ float xi[3];
  const int tid = threadIdx.x;
  for (int i=tid; i<HID*64; i+=128) W2s[i] = W2[i];
  float w0=0.f,w1=0.f,w2=0.f;
  if (tid < 64){ w0=W1[tid*3]; w1=W1[tid*3+1]; w2=W1[tid*3+2]; }
  __syncthreads();
  for (int rr=0; rr<128; rr++){
    size_t row = (size_t)blockIdx.x*128 + rr;
    if (tid < 3) xi[tid] = inp[row*3 + tid];
    __syncthreads();
    if (tid < 64) e[tid] = fmaxf(xi[0]*w0 + xi[1]*w1 + xi[2]*w2, 0.f);
    __syncthreads();
    float acc = 0.f;
    const float* wr = W2s + tid*64;
#pragma unroll 8
    for (int n=0;n<64;n++) acc += e[n]*wr[n];
    x0[row*HID + tid] = (__bf16)fmaxf(acc, 0.f);
    __syncthreads();
  }
}

// ---------------------------------------------------------------- encoder (layer-major)
// LDS: [0, 256K) weights | [256K, +8K) Xb0 | +8K Xb1 | +8K Hb   => 286720 B
#define ENC_XOFF  ((unsigned)WBYTES)
#define ENC_HOFF  ((unsigned)(WBYTES + 2*XBYTES))
__global__ __launch_bounds__(256) void enc_layer_kernel(
    const __bf16* __restrict__ xin, __bf16* __restrict__ xout,
    const __bf16* __restrict__ Wc, float* __restrict__ cOut){
  extern __shared__ char smem[];
  __bf16* Wlds = (__bf16*)smem;
  __bf16* Xb0  = (__bf16*)(smem + ENC_XOFF);
  __bf16* Xb1  = Xb0 + BT*HID;
  __bf16* Hb   = (__bf16*)(smem + ENC_HOFF);
  const int tid = threadIdx.x, lane = tid & 31, wv = tid >> 5, j = wv;
  const int bg = blockIdx.x;

#if HAVE_TDM
  if (wv == 0){
    // weights (512 rows x 256 cols bf16), then x_0 tile (32 rows x 128, row stride T*H)
    tdm_load_2d(0u, Wc, KTOT, GATES, KTOT);
    tdm_load_2d(ENC_XOFF, xin + ((size_t)(bg*BT)*TLEN + 0)*HID, HID, BT, TLEN*HID);
  }
#else
  {
    const uint4* wsrc = (const uint4*)Wc;
    uint4* wdst = (uint4*)Wlds;
    for (int i=tid; i<(int)(WBYTES/16); i+=256) wdst[i] = wsrc[i];
    for (int q=tid; q<(BT*HID)/8; q+=256){
      int row = q >> 4, cq = q & 15;
      ((uint4*)Xb0)[q] = *(const uint4*)(xin + ((size_t)(bg*BT+row)*TLEN + 0)*HID + cq*8);
    }
  }
#endif
  for (int i=tid; i<(BT*HID)/8; i+=256) ((uint4*)Hb)[i] = make_uint4(0u,0u,0u,0u);
  v8f cst[2];
#pragma unroll
  for (int m=0;m<2;m++)
#pragma unroll
    for (int r=0;r<8;r++) cst[m][r] = 0.f;

  const int nn = lane & 15, hf = lane >> 4;
  for (int t=0; t<TLEN; t++){
#if HAVE_TDM
    if (wv == 0){
      if (t+1 < TLEN){
        tdm_load_2d(ENC_XOFF + (unsigned)(((t+1)&1)*XBYTES),
                    xin + ((size_t)(bg*BT)*TLEN + (t+1))*HID, HID, BT, TLEN*HID);
        __builtin_amdgcn_s_wait_tensorcnt(1);   // tile t (and weights) resident
      } else {
        __builtin_amdgcn_s_wait_tensorcnt(0);
      }
    }
#else
    if (t > 0){
      __bf16* Xcur = (t&1) ? Xb1 : Xb0;
      for (int q=tid; q<(BT*HID)/8; q+=256){
        int row = q >> 4, cq = q & 15;
        ((uint4*)Xcur)[q] = *(const uint4*)(xin + ((size_t)(bg*BT+row)*TLEN + t)*HID + cq*8);
      }
    }
#endif
    __syncthreads();                            // x_t + previous h writes visible
    const __bf16* Xb = (t&1) ? Xb1 : Xb0;
    v8f acc[2][4];
    lstm_cell_gemm(Wlds, Xb, Hb, j, lane, acc);
    __syncthreads();                            // all Hb reads complete
#pragma unroll
    for (int m=0;m<2;m++){
#pragma unroll
      for (int r=0;r<8;r++){
        float iv = sigm(acc[m][0][r]);
        float fv = sigm(acc[m][1][r]);
        float gv = tanh_f(acc[m][2][r]);
        float ov = sigm(acc[m][3][r]);
        float c  = fv*cst[m][r] + iv*gv;
        cst[m][r] = c;
        float h  = ov*tanh_f(c);
        int M = m*16 + r + hf*8;
        Hb[M*HID + j*16 + nn] = (__bf16)h;
        if (xout)
          xout[((size_t)(bg*BT+M)*TLEN + t)*HID + j*16 + nn] = (__bf16)h;
      }
    }
  }
  if (cOut){
    __syncthreads();
#pragma unroll
    for (int m=0;m<2;m++)
#pragma unroll
      for (int r=0;r<8;r++){
        int M = m*16 + r + hf*8;
        cOut[(size_t)(bg*BT+M)*HID + j*16 + nn] = cst[m][r];
      }
  }
}

// ---------------------------------------------------------------- decoder (5 steps x 4 layers)
// LDS: [0,256K) weights | 4 x 8K layer-h | 8K zeros  => 303104 B
#define DEC_HOFF ((unsigned)WBYTES)
__global__ __launch_bounds__(256) void dec_kernel(
    const __bf16* __restrict__ WcD, const float* __restrict__ cLast,
    const float* __restrict__ ln_g, const float* __restrict__ ln_b,
    const float* __restrict__ outW, const float* __restrict__ outb,
    float* __restrict__ out){
  extern __shared__ char smem[];
  __bf16* Wlds = (__bf16*)smem;
  __bf16* Hl   = (__bf16*)(smem + DEC_HOFF);
  __bf16* Zb   = Hl + 4*BT*HID;
  const int tid = threadIdx.x, lane = tid & 31, wv = tid >> 5, j = wv;
  const int bg = blockIdx.x, d = blockIdx.y;
  for (int i=tid; i<(5*BT*HID)/8; i+=256) ((uint4*)Hl)[i] = make_uint4(0u,0u,0u,0u);
  const int nn = lane & 15, hf = lane >> 4;
  v8f cst[4][2];
#pragma unroll
  for (int l=0;l<4;l++)
#pragma unroll
    for (int m=0;m<2;m++)
#pragma unroll
      for (int r=0;r<8;r++) cst[l][m][r] = 0.f;
#pragma unroll
  for (int m=0;m<2;m++)                         // layer-0 cell = encoder final cell
#pragma unroll
    for (int r=0;r<8;r++){
      int M = m*16 + r + hf*8;
      cst[0][m][r] = cLast[(size_t)(bg*BT+M)*HID + j*16 + nn];
    }
  __syncthreads();

  for (int s=0; s<5; s++){
    for (int l=0; l<4; l++){
#if HAVE_TDM
      if (wv == 0){
        tdm_load_2d(0u, WcD + ((size_t)(d*4+l))*WELEM, KTOT, GATES, KTOT);
        __builtin_amdgcn_s_wait_tensorcnt(0);
      }
#else
      {
        const uint4* wsrc = (const uint4*)(WcD + ((size_t)(d*4+l))*WELEM);
        uint4* wdst = (uint4*)Wlds;
        for (int i=tid; i<(int)(WBYTES/16); i+=256) wdst[i] = wsrc[i];
      }
#endif
      if (l < 3) __builtin_prefetch(WcD + ((size_t)(d*4+l+1))*WELEM + tid*64, 0, 1);
      __syncthreads();
      const __bf16* Xb = (l==0) ? ((s==0) ? Zb : (Hl + 3*BT*HID)) : (Hl + (l-1)*BT*HID);
      __bf16* Hb = Hl + l*BT*HID;
      v8f acc[2][4];
      lstm_cell_gemm(Wlds, Xb, Hb, j, lane, acc);
      __syncthreads();
#pragma unroll
      for (int m=0;m<2;m++){
#pragma unroll
        for (int r=0;r<8;r++){
          float iv = sigm(acc[m][0][r]);
          float fv = sigm(acc[m][1][r]);
          float gv = tanh_f(acc[m][2][r]);
          float ov = sigm(acc[m][3][r]);
          float c  = fv*cst[l][m][r] + iv*gv;
          cst[l][m][r] = c;
          float h  = ov*tanh_f(c);
          int M = m*16 + r + hf*8;
          Hb[M*HID + j*16 + nn] = (__bf16)h;
        }
      }
      __syncthreads();
    }
    // LayerNorm + Linear(128->3) on top output, one row per thread
    if (tid < BT){
      const __bf16* hr = Hl + 3*BT*HID + tid*HID;
      float mu = 0.f;
      for (int n=0;n<HID;n++) mu += (float)hr[n];
      mu *= (1.0f/HID);
      float var = 0.f;
      for (int n=0;n<HID;n++){ float dx = (float)hr[n] - mu; var += dx*dx; }
      var *= (1.0f/HID);
      float rstd = rsqrtf(var + 1e-5f);
#pragma unroll
      for (int jj=0;jj<3;jj++){
        float a = outb[d*3+jj];
        const float* wr = outW + (size_t)(d*3+jj)*HID;
        for (int n=0;n<HID;n++){
          float xn = ((float)hr[n]-mu)*rstd*ln_g[d*HID+n] + ln_b[d*HID+n];
          a += xn*wr[n];
        }
        out[(((size_t)(bg*BT+tid)*5 + s)*2 + d)*3 + jj] = a;
      }
    }
  }
}

// ---------------------------------------------------------------- host
extern "C" void kernel_launch(void* const* d_in, const int* in_sizes, int n_in,
                              void* d_out, int out_size, void* d_ws, size_t ws_size,
                              hipStream_t stream){
  const float* inp    = (const float*)d_in[0];
  const float* emb1W  = (const float*)d_in[1];
  const float* emb2W  = (const float*)d_in[2];
  const float* encWih = (const float*)d_in[3];
  const float* encWhh = (const float*)d_in[4];
  const float* decWih = (const float*)d_in[5];
  const float* decWhh = (const float*)d_in[6];
  const float* ln_g   = (const float*)d_in[7];
  const float* ln_b   = (const float*)d_in[8];
  const float* outW   = (const float*)d_in[9];
  const float* outb   = (const float*)d_in[10];
  (void)in_sizes; (void)n_in; (void)out_size; (void)ws_size;

  const size_t xElems = (size_t)BATCH*TLEN*HID;
  __bf16* xA  = (__bf16*)d_ws;
  __bf16* xB  = xA + xElems;
  __bf16* WcE = xB + xElems;
  __bf16* WcD = WcE + 4*(size_t)WELEM;
  float*  cL  = (float*)(WcD + 8*(size_t)WELEM);

  prep_weights_kernel<<<(4*WELEM + 8*WELEM)/256, 256, 0, stream>>>(
      encWih, encWhh, decWih, decWhh, WcE, WcD);
  embed_kernel<<<(BATCH*TLEN)/128, 128, 0, stream>>>(inp, emb1W, emb2W, xA);

  const size_t encLds = WBYTES + 3*XBYTES;                      // 286720 B
  enc_layer_kernel<<<BATCH/BT, 256, encLds, stream>>>(xA, xB, WcE + 0*(size_t)WELEM, nullptr);
  enc_layer_kernel<<<BATCH/BT, 256, encLds, stream>>>(xB, xA, WcE + 1*(size_t)WELEM, nullptr);
  enc_layer_kernel<<<BATCH/BT, 256, encLds, stream>>>(xA, xB, WcE + 2*(size_t)WELEM, nullptr);
  enc_layer_kernel<<<BATCH/BT, 256, encLds, stream>>>(xB, nullptr, WcE + 3*(size_t)WELEM, cL);

  const size_t decLds = WBYTES + 5*XBYTES;                      // 303104 B
  dec_kernel<<<dim3(BATCH/BT, 2), 256, decLds, stream>>>(
      WcD, cL, ln_g, ln_b, outW, outb, (float*)d_out);
}
